// FastAttention_70592082477549
// MI455X (gfx1250) — matrix operand order repaired and checked by reading, not compiled, not run on hardware
//
#include <hip/hip_runtime.h>
#include <math.h>

// Performer (FAVOR+) attention, fused for MI455X (gfx1250, wave32, WMMA).
// B=4 H=16 N=4096 D=64 M=256.

typedef __bf16 bf16;
typedef bf16  v16bf __attribute__((ext_vector_type(16)));
typedef float v8f   __attribute__((ext_vector_type(8)));

#define B_  4
#define H_  16
#define N_  4096
#define D_  64
#define M_  256
#define BH_ (B_*H_)
#define TN  128
#define NCHUNK (N_/TN)

#define DATA_NORM 0.35355339059327373f  /* 64^-0.25 */
#define RATIO     0.0625f               /* 256^-0.5 */
#define EPS       1e-4f

__device__ __forceinline__ unsigned short f2bf(float f) {
  union { float f; unsigned u; } x; x.f = f;
  unsigned r = x.u + 0x7FFFu + ((x.u >> 16) & 1u);   // round-to-nearest-even
  return (unsigned short)(r >> 16);
}
__device__ __forceinline__ float bf2f(unsigned short b) {
  union { unsigned u; float f; } x; x.u = ((unsigned)b) << 16; return x.f;
}
// order-preserving float<->uint for atomicMax on floats
__device__ __forceinline__ unsigned f2ord(float f) {
  union { float f; unsigned u; } x; x.f = f;
  return (x.u & 0x80000000u) ? ~x.u : (x.u | 0x80000000u);
}
__device__ __forceinline__ float ord2f(unsigned u) {
  unsigned v = (u & 0x80000000u) ? (u & 0x7FFFFFFFu) : ~u;
  union { unsigned u; float f; } x; x.u = v; return x.f;
}

// A fragment (16x32 bf16, MxK): LDS row-major [row][ld], K contiguous.
// lane: row = row0+(l&15); lower half-lanes K=k0+[0..7,16..23], upper K=k0+8+[...]
__device__ __forceinline__ v16bf frag_a(const unsigned short* base, int ld,
                                        int row0, int k0, int lane) {
  int row = row0 + (lane & 15);
  int kb  = k0 + ((lane >> 4) << 3);
  const unsigned short* p = base + row * ld + kb;
  union { v16bf v; uint4 q[2]; } u;
  u.q[0] = *(const uint4*)(p);
  u.q[1] = *(const uint4*)(p + 16);
  return u.v;
}
// B fragment (32x16 bf16, KxN): B[k][n] stored transposed: baseT[n*ld + k]
// lane: col = col0+(l&15); lower half-lanes K=k0+[0..7,16..23], upper K=k0+8+[...]
__device__ __forceinline__ v16bf frag_b(const unsigned short* baseT, int ld,
                                        int col0, int k0, int lane) {
  int col = col0 + (lane & 15);
  int kb  = k0 + ((lane >> 4) << 3);
  const unsigned short* p = baseT + col * ld + kb;
  union { v16bf v; uint4 q[2]; } u;
  u.q[0] = *(const uint4*)(p);
  u.q[1] = *(const uint4*)(p + 16);
  return u.v;
}

__device__ __forceinline__ v8f wmma_bf16(v16bf a, v16bf b, v8f c) {
  return __builtin_amdgcn_wmma_f32_16x16x32_bf16(false, a, false, b,
                                                 (short)0, c, false, false);
}

// stage projection matrix [M_][D_] f32 -> LDS bf16 (same row-major layout;
// serves as transposed-B storage since proj[m][k] == B[k][m])
__device__ __forceinline__ void stage_proj(const float* proj,
                                           unsigned short* projLds, int tid) {
  const float4* p4 = (const float4*)proj;
  #pragma unroll
  for (int j = 0; j < 16; ++j) {
    int q = tid + j * 256;            // 4096 float4
    float4 w = p4[q];
    int o = q * 4;
    projLds[o+0] = f2bf(w.x); projLds[o+1] = f2bf(w.y);
    projLds[o+2] = f2bf(w.z); projLds[o+3] = f2bf(w.w);
  }
}

// stage a TN x D_ chunk of q/k, pre-scaled by data_normalizer, to LDS bf16.
// Optionally accumulates per-row sum of squares of the scaled data (for diag).
__device__ __forceinline__ void stage_x(const float* xg, unsigned short* xLds,
                                        float* diagLds, int tid) {
  const float4* p4 = (const float4*)xg;
  #pragma unroll
  for (int j = 0; j < 8; ++j) {
    int q = tid + j * 256;            // 2048 float4 = TN*64/4
    float4 w = p4[q];
    float a = w.x * DATA_NORM, b = w.y * DATA_NORM;
    float c = w.z * DATA_NORM, d = w.w * DATA_NORM;
    int o = q * 4;
    xLds[o+0] = f2bf(a); xLds[o+1] = f2bf(b);
    xLds[o+2] = f2bf(c); xLds[o+3] = f2bf(d);
    if (diagLds) atomicAdd(&diagLds[o >> 6], a*a + b*b + c*c + d*d);
  }
}

__global__ void perf_init(unsigned* kmaxWs, float* ksumWs, float* ctxWs) {
  int i = blockIdx.x * blockDim.x + threadIdx.x;
  if (i < BH_)      kmaxWs[i] = 0x007FFFFFu;   // f2ord(-inf)
  if (i < BH_ * M_) ksumWs[i] = 0.f;
  for (size_t j = i; j < (size_t)BH_ * M_ * D_; j += (size_t)gridDim.x * blockDim.x)
    ctxWs[j] = 0.f;
}

// Pass 1: global max over (n, m) of k's data_dash, per (b,h).
__global__ __launch_bounds__(256) void kmax_kernel(const float* __restrict__ k,
                                                   const float* __restrict__ proj,
                                                   unsigned* kmaxWs) {
  __shared__ __align__(16) unsigned short projLds[M_ * D_];
  __shared__ __align__(16) unsigned short xLds[TN * D_];
  __shared__ float red[256];
  int tid = threadIdx.x, lane = tid & 31, wave = tid >> 5;
  int bh = blockIdx.x, n0 = blockIdx.y * TN;
  stage_proj(proj, projLds, tid);
  stage_x(k + ((size_t)bh * N_ + n0) * D_, xLds, nullptr, tid);
  __syncthreads();
  int r0 = wave * 16;
  v16bf a0 = frag_a(xLds, D_, r0, 0, lane);
  v16bf a1 = frag_a(xLds, D_, r0, 32, lane);
  float mx = -3.4e38f;
  for (int ct = 0; ct < 16; ++ct) {
    v16bf b0 = frag_b(projLds, D_, ct * 16, 0, lane);
    v16bf b1 = frag_b(projLds, D_, ct * 16, 32, lane);
    v8f acc = {};
    acc = wmma_bf16(a0, b0, acc);
    acc = wmma_bf16(a1, b1, acc);
    #pragma unroll
    for (int j = 0; j < 8; ++j) mx = fmaxf(mx, acc[j]);
  }
  red[tid] = mx;
  __syncthreads();
  for (int s = 128; s > 0; s >>= 1) {
    if (tid < s) red[tid] = fmaxf(red[tid], red[tid + s]);
    __syncthreads();
  }
  if (tid == 0) atomicMax(&kmaxWs[bh], f2ord(red[0]));
}

// Pass 2: kp = ratio*(exp(dd - diag - kmax)+eps); ksum += col-sums(kp);
//         ctx += kp^T @ v. kp lives only in LDS.
__global__ __launch_bounds__(256) void context_kernel(const float* __restrict__ k,
                                                      const float* __restrict__ v,
                                                      const float* __restrict__ proj,
                                                      const unsigned* __restrict__ kmaxWs,
                                                      float* ksumWs, float* ctxWs) {
  __shared__ __align__(16) unsigned short projLds[M_ * D_];
  __shared__ __align__(16) unsigned short xLds[TN * D_];
  __shared__ __align__(16) unsigned short vT[D_ * TN];    // [e][n_local]
  __shared__ __align__(16) unsigned short kpT[M_ * TN];   // [m][n_local]
  __shared__ float diagLds[TN];
  int tid = threadIdx.x, lane = tid & 31, wave = tid >> 5;
  int bh = blockIdx.x, n0 = blockIdx.y * TN;
  if (tid < TN) diagLds[tid] = 0.f;
  __syncthreads();
  stage_proj(proj, projLds, tid);
  stage_x(k + ((size_t)bh * N_ + n0) * D_, xLds, diagLds, tid);
  {   // stage v transposed: vT[e][n]
    const float4* p4 = (const float4*)(v + ((size_t)bh * N_ + n0) * D_);
    #pragma unroll
    for (int j = 0; j < 8; ++j) {
      int q = tid + j * 256;
      float4 w = p4[q];
      int lin = q * 4, row = lin >> 6, e0 = lin & 63;
      vT[(e0 + 0) * TN + row] = f2bf(w.x);
      vT[(e0 + 1) * TN + row] = f2bf(w.y);
      vT[(e0 + 2) * TN + row] = f2bf(w.z);
      vT[(e0 + 3) * TN + row] = f2bf(w.w);
    }
  }
  __syncthreads();
  float kmax = ord2f(kmaxWs[bh]);
  int r0 = wave * 16, hi = lane >> 4, l15 = lane & 15;
  v16bf a0 = frag_a(xLds, D_, r0, 0, lane);
  v16bf a1 = frag_a(xLds, D_, r0, 32, lane);
  for (int ct = 0; ct < 16; ++ct) {
    v16bf b0 = frag_b(projLds, D_, ct * 16, 0, lane);
    v16bf b1 = frag_b(projLds, D_, ct * 16, 32, lane);
    v8f acc = {};
    acc = wmma_bf16(a0, b0, acc);
    acc = wmma_bf16(a1, b1, acc);
    int m = ct * 16 + l15;
    #pragma unroll
    for (int j = 0; j < 8; j += 2) {       // rows j, j+1 are adjacent -> pack b32
      int r = r0 + j + 8 * hi;
      float q0 = RATIO * (expf(acc[j]     - 0.5f * diagLds[r]     - kmax) + EPS);
      float q1 = RATIO * (expf(acc[j + 1] - 0.5f * diagLds[r + 1] - kmax) + EPS);
      unsigned pack = (unsigned)f2bf(q0) | ((unsigned)f2bf(q1) << 16);
      *(unsigned*)&kpT[m * TN + r] = pack;
    }
  }
  __syncthreads();
  {   // ksum: one m per thread
    float s = 0.f;
    const unsigned* p = (const unsigned*)&kpT[tid * TN];
    for (int j = 0; j < TN / 2; ++j) {
      unsigned u = p[j];
      s += bf2f((unsigned short)(u & 0xFFFFu)) + bf2f((unsigned short)(u >> 16));
    }
    atomicAdd(&ksumWs[bh * M_ + tid], s);
  }
  // GEMM2: ctx[m][e] += sum_n kpT[m][n] * v[n][e]   (A=kpT [256xTN], B=v [TNx64])
  int mt0 = wave * 2;
  v8f zero = {};
  v8f accs[2][4];
  #pragma unroll
  for (int i = 0; i < 2; ++i)
    #pragma unroll
    for (int et = 0; et < 4; ++et) accs[i][et] = zero;
  for (int ks = 0; ks < 4; ++ks) {
    v16bf aA = frag_a(kpT, TN, mt0 * 16,      ks * 32, lane);
    v16bf aB = frag_a(kpT, TN, mt0 * 16 + 16, ks * 32, lane);
    #pragma unroll
    for (int et = 0; et < 4; ++et) {
      v16bf b = frag_b(vT, TN, et * 16, ks * 32, lane);
      accs[0][et] = wmma_bf16(aA, b, accs[0][et]);
      accs[1][et] = wmma_bf16(aB, b, accs[1][et]);
    }
  }
  #pragma unroll
  for (int i = 0; i < 2; ++i)
    #pragma unroll
    for (int et = 0; et < 4; ++et)
      #pragma unroll
      for (int j = 0; j < 8; ++j) {
        int m = (mt0 + i) * 16 + j + 8 * hi;
        int e = et * 16 + l15;
        atomicAdd(&ctxWs[((size_t)bh * M_ + m) * D_ + e], accs[i][et][j]);
      }
}

// Pass 3: qp (per-row max), denom = qp . ksum, out = (qp @ ctx) / denom
__global__ __launch_bounds__(256) void output_kernel(const float* __restrict__ q,
                                                     const float* __restrict__ proj,
                                                     const float* __restrict__ ksumWs,
                                                     const float* __restrict__ ctxWs,
                                                     float* __restrict__ out) {
  __shared__ __align__(16) unsigned short projLds[M_ * D_];
  __shared__ __align__(16) unsigned short xLds[TN * D_];
  __shared__ __align__(16) unsigned short qpLds[TN * M_];  // [n][m]
  __shared__ __align__(16) unsigned short ctxT[D_ * M_];   // [e][m]
  __shared__ float ksumLds[M_];
  __shared__ float diagLds[TN];
  __shared__ float dinvLds[TN];
  int tid = threadIdx.x, lane = tid & 31, wave = tid >> 5;
  int bh = blockIdx.x, n0 = blockIdx.y * TN;
  if (tid < TN) diagLds[tid] = 0.f;
  __syncthreads();
  stage_proj(proj, projLds, tid);
  stage_x(q + ((size_t)bh * N_ + n0) * D_, xLds, diagLds, tid);
  ksumLds[tid] = ksumWs[bh * M_ + tid];
  {   // stage context transposed: ctxT[e][m]
    const float4* p4 = (const float4*)(ctxWs + (size_t)bh * M_ * D_);
    #pragma unroll
    for (int j = 0; j < 16; ++j) {
      int qq = tid + j * 256;        // 4096 float4
      float4 w = p4[qq];
      int lin = qq * 4, m = lin >> 6, e0 = lin & 63;
      ctxT[(e0 + 0) * M_ + m] = f2bf(w.x);
      ctxT[(e0 + 1) * M_ + m] = f2bf(w.y);
      ctxT[(e0 + 2) * M_ + m] = f2bf(w.z);
      ctxT[(e0 + 3) * M_ + m] = f2bf(w.w);
    }
  }
  __syncthreads();
  int r0 = wave * 16, hi = lane >> 4, l15 = lane & 15;
  v16bf a0 = frag_a(xLds, D_, r0, 0, lane);
  v16bf a1 = frag_a(xLds, D_, r0, 32, lane);
  v8f zero = {};
  v8f accs[16];
  #pragma unroll
  for (int ct = 0; ct < 16; ++ct) accs[ct] = zero;
  #pragma unroll
  for (int ct = 0; ct < 16; ++ct) {
    v16bf b0 = frag_b(projLds, D_, ct * 16, 0, lane);
    v16bf b1 = frag_b(projLds, D_, ct * 16, 32, lane);
    accs[ct] = wmma_bf16(a0, b0, accs[ct]);
    accs[ct] = wmma_bf16(a1, b1, accs[ct]);
  }
  // per-row max of data_dash (row r0 + j + 8*hi; cols across 16-lane half)
  float mx[8];
  #pragma unroll
  for (int j = 0; j < 8; ++j) {
    float m = accs[0][j];
    #pragma unroll
    for (int ct = 1; ct < 16; ++ct) m = fmaxf(m, accs[ct][j]);
    for (int d = 1; d < 16; d <<= 1) m = fmaxf(m, __shfl_xor(m, d, 32));
    mx[j] = m;
  }
  // qp + denominator partials
  float den[8];
  #pragma unroll
  for (int j = 0; j < 8; ++j) den[j] = 0.f;
  #pragma unroll
  for (int ct = 0; ct < 16; ++ct) {
    int m = ct * 16 + l15;
    float s = ksumLds[m];
    #pragma unroll
    for (int j = 0; j < 8; ++j) {
      int r = r0 + j + 8 * hi;
      float qv = RATIO * (expf(accs[ct][j] - 0.5f * diagLds[r] - mx[j]) + EPS);
      den[j] += qv * s;
      qpLds[r * M_ + m] = f2bf(qv);
    }
  }
  #pragma unroll
  for (int j = 0; j < 8; ++j) {
    float d = den[j];
    for (int t = 1; t < 16; t <<= 1) d += __shfl_xor(d, t, 32);
    if (l15 == 0) dinvLds[r0 + j + 8 * hi] = 1.f / d;
  }
  __syncthreads();
  // GEMM2: out_tile = qp [TN x 256] @ ctx [256 x 64]
  v8f oacc[4];
  #pragma unroll
  for (int et = 0; et < 4; ++et) oacc[et] = zero;
  for (int ks = 0; ks < 8; ++ks) {
    v16bf a = frag_a(qpLds, M_, r0, ks * 32, lane);
    #pragma unroll
    for (int et = 0; et < 4; ++et) {
      v16bf b = frag_b(ctxT, M_, et * 16, ks * 32, lane);
      oacc[et] = wmma_bf16(a, b, oacc[et]);
    }
  }
  float* og = out + ((size_t)bh * N_ + n0) * D_;
  #pragma unroll
  for (int et = 0; et < 4; ++et)
    #pragma unroll
    for (int j = 0; j < 8; ++j) {
      int r = r0 + j + 8 * hi;
      int e = et * 16 + l15;
      og[(size_t)r * D_ + e] = oacc[et][j] * dinvLds[r];
    }
}

extern "C" void kernel_launch(void* const* d_in, const int* in_sizes, int n_in,
                              void* d_out, int out_size, void* d_ws, size_t ws_size,
                              hipStream_t stream) {
  const float* q    = (const float*)d_in[0];
  const float* k    = (const float*)d_in[1];
  const float* v    = (const float*)d_in[2];
  const float* proj = (const float*)d_in[3];

  unsigned* kmaxWs = (unsigned*)d_ws;
  float*    ksumWs = (float*)((char*)d_ws + 256);       // 16B-aligned
  float*    ctxWs  = ksumWs + BH_ * M_;                 // 64*256*64 f32 = 4MB
  float*    out    = (float*)d_out;

  perf_init<<<1024, 256, 0, stream>>>(kmaxWs, ksumWs, ctxWs);
  dim3 grid(BH_, NCHUNK);
  kmax_kernel<<<grid, 256, 0, stream>>>(k, proj, kmaxWs);
  context_kernel<<<grid, 256, 0, stream>>>(k, v, proj, kmaxWs, ksumWs, ctxWs);
  output_kernel<<<grid, 256, 0, stream>>>(q, proj, ksumWs, ctxWs, out);
}
